// AttentionBlock_47047071761080
// MI455X (gfx1250) — matrix operand order repaired and checked by reading, not compiled, not run on hardware
//
#include <hip/hip_runtime.h>
#include <hip/hip_bf16.h>
#include <math.h>

// ---------------------------------------------------------------------------
// Problem constants: B=8, C=512, H=W=32 -> L=1024, N_HEADS=8, CH=64, GROUPS=32
// ---------------------------------------------------------------------------
#define BB 8
#define CC 512
#define LL 1024
#define NH 8
#define CH 64
#define NGROUP 32
// scale = 1/sqrt(sqrt(64)) applied to BOTH q and k (as in reference)
#define QK_SCALE 0.35355339059327373f

typedef _Float16 half8 __attribute__((ext_vector_type(8)));
typedef _Float16 v16h  __attribute__((ext_vector_type(16)));
typedef float    v8f   __attribute__((ext_vector_type(8)));

// ---------------------------------------------------------------------------
// WMMA wrapper: D = A(16x32 f16) * B(32x16 f16) + C(16x16 f32)
// ---------------------------------------------------------------------------
__device__ __forceinline__ v8f wmma_f16(v16h a, v16h b, v8f c) {
    return __builtin_amdgcn_wmma_f32_16x16x32_f16(
        false, a, false, b, (short)0, c, false, false);
}

// ---------------------------------------------------------------------------
// Fragment loaders.
// A (16x32): lane = m + 16*g ; element j -> K = (j/8)*16 + g*8 + (j%8)
// B (32x16): lane = n + 16*g ; element j -> K = (j/8)*16 + g*8 + (j%8)
// C (16x16 f32): lane = n + 16*g ; reg r -> M = r + 8*g
// ---------------------------------------------------------------------------

// A row-major: (m,k) at base[m*ld + k]  -> two contiguous b128 loads per lane
__device__ __forceinline__ v16h ld_a_row(const _Float16* base, int ld) {
    int lane = threadIdx.x & 31;
    int m = lane & 15, g = lane >> 4;
    const _Float16* p = base + (size_t)m * ld + g * 8;
    half8 lo = *(const half8*)(p);
    half8 hi = *(const half8*)(p + 16);
    v16h f;
#pragma unroll
    for (int j = 0; j < 8; ++j) { f[j] = lo[j]; f[j + 8] = hi[j]; }
    return f;
}

// B row-major: (k,n) at base[k*ld + n]  -> lane = n + 16*g, element j -> K = (j/8)*16 + g*8 + (j%8)
__device__ __forceinline__ v16h ld_b_row(const _Float16* base, int ld) {
    int lane = threadIdx.x & 31;
    int n = lane & 15, g = lane >> 4;
    const _Float16* p = base + n + (size_t)(g * 8) * ld;
    v16h f;
#pragma unroll
    for (int j = 0; j < 8; ++j) { f[j] = p[(size_t)j * ld]; f[j + 8] = p[(size_t)(16 + j) * ld]; }
    return f;
}

// ---------------------------------------------------------------------------
// Async global -> LDS staging (CDNA5 path, tracked by ASYNCcnt)
// ---------------------------------------------------------------------------
__device__ __forceinline__ void async_copy_b128(const _Float16* gsrc, _Float16* ldst) {
    unsigned long long ga = (unsigned long long)(size_t)gsrc;
    unsigned la = (unsigned)(size_t)ldst; // low 32 bits = LDS offset
    asm volatile("global_load_async_to_lds_b128 %0, %1, off"
                 :: "v"(la), "v"(ga) : "memory");
}
__device__ __forceinline__ void wait_async() {
    asm volatile("s_wait_asynccnt 0x0" ::: "memory");
}

// ---------------------------------------------------------------------------
// Kernel 1: convert qkv_w (1536x512) and proj_w (512x512) to f16
// ---------------------------------------------------------------------------
__global__ void cvt_weights_kernel(const float* __restrict__ qkvw,
                                   const float* __restrict__ projw,
                                   _Float16* __restrict__ qkvh,
                                   _Float16* __restrict__ projh) {
    int i = blockIdx.x * blockDim.x + threadIdx.x;
    if (i < 3 * CC * CC) qkvh[i] = (_Float16)qkvw[i];
    if (i < CC * CC)     projh[i] = (_Float16)projw[i];
}

// ---------------------------------------------------------------------------
// Kernel 2: GroupNorm over (B, C, L). One block per (b, group).
// ---------------------------------------------------------------------------
__global__ void groupnorm_kernel(const float* __restrict__ x,
                                 const float* __restrict__ w,
                                 const float* __restrict__ bias,
                                 _Float16* __restrict__ hn) {
    const int GN_ELEMS = (CC / NGROUP) * LL; // 16*1024
    int b = blockIdx.x >> 5;
    int g = blockIdx.x & 31;
    const float* xp = x + (size_t)b * CC * LL + (size_t)g * GN_ELEMS;
    int tid = threadIdx.x;

    float s = 0.f, s2 = 0.f;
    for (int i = tid; i < GN_ELEMS; i += 256) {
        float v = xp[i];
        s += v; s2 += v * v;
    }
    __shared__ float rs[256], rs2[256];
    rs[tid] = s; rs2[tid] = s2;
    __syncthreads();
    for (int off = 128; off; off >>= 1) {
        if (tid < off) { rs[tid] += rs[tid + off]; rs2[tid] += rs2[tid + off]; }
        __syncthreads();
    }
    float mean = rs[0] * (1.0f / GN_ELEMS);
    float var  = rs2[0] * (1.0f / GN_ELEMS) - mean * mean;
    float rstd = rsqrtf(var + 1e-5f);

    _Float16* hp = hn + (size_t)b * CC * LL + (size_t)g * GN_ELEMS;
    for (int i = tid; i < GN_ELEMS; i += 256) {
        int c = g * 16 + (i >> 10);
        hp[i] = (_Float16)((xp[i] - mean) * rstd * w[c] + bias[c]);
    }
}

// ---------------------------------------------------------------------------
// Kernel 3: QKV GEMM per batch: [1536 x 512] x [512 x 1024].
// 8 waves/block, one 16x16 output tile each (block = 128 rows x 16 cols).
// The shared 512x16 B panel (16 KB) is staged once in LDS with
// global_load_async_to_lds_b128; waves then read it with strided 16-bit LDS loads.
// Epilogue: bias + route into qT (b*NH+h, L, CH) scaled, k (b*NH+h, CH, L)
// scaled, vT (b*NH+h, L, CH).
// ---------------------------------------------------------------------------
__global__ void qkv_gemm_kernel(const _Float16* __restrict__ W,
                                const _Float16* __restrict__ hn,
                                const float* __restrict__ bias,
                                _Float16* __restrict__ qT,
                                _Float16* __restrict__ kh,
                                _Float16* __restrict__ vT) {
    __shared__ _Float16 Bs[CC * 16]; // 16 KB B panel
    int tid  = threadIdx.x;
    int wave = tid >> 5;
    int mrow = blockIdx.y * 128 + wave * 16;
    int ncol = blockIdx.x * 16;
    int b = blockIdx.z;
    const _Float16* Bb = hn + (size_t)b * CC * LL + ncol;

    // --- async-stage the full K panel of B: 512 rows x 32 bytes ---
#pragma unroll
    for (int i = 0; i < 4; ++i) {
        int t = tid + 256 * i;        // 0..1023 transfers of 16B
        int row = t >> 1, h = t & 1;
        async_copy_b128(Bb + (size_t)row * LL + h * 8, Bs + row * 16 + h * 8);
    }
    wait_async();
    __syncthreads();

    const _Float16* Arow = W + (size_t)mrow * CC;
    v8f acc = {};
    for (int k0 = 0; k0 < CC; k0 += 32) {
        __builtin_prefetch(Arow + k0 + 64, 0, 0);
        v16h a  = ld_a_row(Arow + k0, CC);
        v16h bf = ld_b_row(Bs + k0 * 16, 16);
        acc = wmma_f16(a, bf, acc);
    }

    int lane = tid & 31;
    int n = lane & 15, g = lane >> 4;
    int col = ncol + n;
#pragma unroll
    for (int r = 0; r < 8; ++r) {
        int orow = mrow + r + 8 * g;
        float v = acc[r] + bias[orow];
        int head = orow / (3 * CH);
        int rr = orow - head * (3 * CH);
        size_t bh = (size_t)(b * NH + head);
        if (rr < CH)                 // q, transposed (l, ch), pre-scaled
            qT[(bh * LL + col) * CH + rr] = (_Float16)(v * QK_SCALE);
        else if (rr < 2 * CH)        // k, (ch, l), pre-scaled
            kh[(bh * CH + (rr - CH)) * LL + col] = (_Float16)(v * QK_SCALE);
        else                         // v, transposed (l, ch)
            vT[(bh * LL + col) * CH + (rr - 2 * CH)] = (_Float16)v;
    }
}

// ---------------------------------------------------------------------------
// Kernel 4: attention per (head-batch, 16-row query tile). One wave/block.
// Phase 1: S[16,1024] via WMMA (q A-fragments hoisted, loaded once).
// Phase 2: row softmax (shuffle reductions), probs written f16 row-major.
// Phase 3: O = P x V^T via WMMA with ds_load_b128 A and strided 16-bit global B.
// Dynamic LDS: 64 KB f32 scores + 32 KB f16 probs = 96 KB.
// ---------------------------------------------------------------------------
__global__ void attn_kernel(const _Float16* __restrict__ qT,
                            const _Float16* __restrict__ kh,
                            const _Float16* __restrict__ vT,
                            _Float16* __restrict__ ah) {
    extern __shared__ float sc[];                 // [16][1024] f32 scores
    _Float16* p16 = (_Float16*)(sc + 16 * LL);    // [16][1024] f16 probs
    int lane = threadIdx.x;
    int t0 = blockIdx.x * 16;
    int bh = blockIdx.y;
    const _Float16* qb = qT + (size_t)bh * LL * CH;  // (l, ch)
    const _Float16* kb = kh + (size_t)bh * CH * LL;  // (ch, l)
    const _Float16* vb = vT + (size_t)bh * LL * CH;  // (l, ch)
    int n = lane & 15, g = lane >> 4;

    // ---- Phase 1: scores; q fragments loaded once, reused for 64 tiles ----
    v16h aq0 = ld_a_row(qb + (size_t)t0 * CH + 0,  CH); // (t, c) rows 0..31 of K
    v16h aq1 = ld_a_row(qb + (size_t)t0 * CH + 32, CH); // (t, c) rows 32..63
    for (int s0 = 0; s0 < LL; s0 += 16) {
        v8f acc = {};
        acc = wmma_f16(aq0, ld_b_row(kb + (size_t)0  * LL + s0, LL), acc);
        acc = wmma_f16(aq1, ld_b_row(kb + (size_t)32 * LL + s0, LL), acc);
#pragma unroll
        for (int r = 0; r < 8; ++r)
            sc[(r + 8 * g) * LL + s0 + n] = acc[r];
    }
    __syncthreads();

    // ---- Phase 2: softmax over s for each of the 16 rows ----
    for (int row = 0; row < 16; ++row) {
        float* pr = sc + row * LL;
        _Float16* pp = p16 + row * LL;
        float mx = -3.4e38f;
        for (int i = lane; i < LL; i += 32) mx = fmaxf(mx, pr[i]);
#pragma unroll
        for (int off = 16; off; off >>= 1) mx = fmaxf(mx, __shfl_xor(mx, off, 32));
        float sum = 0.f;
        for (int i = lane; i < LL; i += 32) {
            float e = __expf(pr[i] - mx);
            pr[i] = e;
            sum += e;
        }
#pragma unroll
        for (int off = 16; off; off >>= 1) sum += __shfl_xor(sum, off, 32);
        float inv = 1.0f / sum;
        for (int i = lane; i < LL; i += 32)
            pp[i] = (_Float16)(pr[i] * inv);
    }
    __syncthreads();

    // ---- Phase 3: O = P x V^T ----
    for (int c0 = 0; c0 < CH; c0 += 16) {
        v8f acc = {};
        for (int kc = 0; kc < LL; kc += 32) {
            v16h a  = ld_a_row(p16 + kc, LL);               // (t,s) f16, ds_b128
            v16h bf = ld_b_row(vb + (size_t)kc * CH + c0, CH); // (s,c) f16, strided
            acc = wmma_f16(a, bf, acc);
        }
#pragma unroll
        for (int r = 0; r < 8; ++r)
            ah[((size_t)bh * CH + c0 + n) * LL + t0 + r + 8 * g] = (_Float16)acc[r];
    }
}

// ---------------------------------------------------------------------------
// Kernel 5: proj GEMM per batch + bias + residual, with async LDS B staging
// ---------------------------------------------------------------------------
__global__ void proj_gemm_kernel(const _Float16* __restrict__ W,
                                 const _Float16* __restrict__ A,
                                 const float* __restrict__ bias,
                                 const float* __restrict__ x,
                                 float* __restrict__ out) {
    __shared__ _Float16 Bs[CC * 16];
    int tid  = threadIdx.x;
    int wave = tid >> 5;
    int mrow = blockIdx.y * 128 + wave * 16;
    int ncol = blockIdx.x * 16;
    int b = blockIdx.z;
    const _Float16* Bb = A + (size_t)b * CC * LL + ncol;

#pragma unroll
    for (int i = 0; i < 4; ++i) {
        int t = tid + 256 * i;
        int row = t >> 1, h = t & 1;
        async_copy_b128(Bb + (size_t)row * LL + h * 8, Bs + row * 16 + h * 8);
    }
    wait_async();
    __syncthreads();

    const _Float16* Arow = W + (size_t)mrow * CC;
    v8f acc = {};
    for (int k0 = 0; k0 < CC; k0 += 32) {
        __builtin_prefetch(Arow + k0 + 64, 0, 0);
        v16h a  = ld_a_row(Arow + k0, CC);
        v16h bf = ld_b_row(Bs + k0 * 16, 16);
        acc = wmma_f16(a, bf, acc);
    }

    int lane = tid & 31;
    int n = lane & 15, g = lane >> 4;
    int col = ncol + n;
#pragma unroll
    for (int r = 0; r < 8; ++r) {
        int orow = mrow + r + 8 * g;
        size_t idx = (size_t)b * CC * LL + (size_t)orow * LL + col;
        out[idx] = x[idx] + acc[r] + bias[orow];
    }
}

// ---------------------------------------------------------------------------
// Launch
// ---------------------------------------------------------------------------
extern "C" void kernel_launch(void* const* d_in, const int* in_sizes, int n_in,
                              void* d_out, int out_size, void* d_ws, size_t ws_size,
                              hipStream_t stream) {
    (void)in_sizes; (void)n_in; (void)out_size; (void)ws_size;
    const float* x      = (const float*)d_in[0];
    const float* gn_w   = (const float*)d_in[1];
    const float* gn_b   = (const float*)d_in[2];
    const float* qkv_w  = (const float*)d_in[3];
    const float* qkv_b  = (const float*)d_in[4];
    const float* proj_w = (const float*)d_in[5];
    const float* proj_b = (const float*)d_in[6];
    float* out = (float*)d_out;

    // Workspace layout (bytes)
    char* ws = (char*)d_ws;
    _Float16* qkvwh  = (_Float16*)(ws);              // 1.5 MB
    _Float16* projwh = (_Float16*)(ws + 1572864);    // 0.5 MB
    _Float16* hn     = (_Float16*)(ws + 2097152);    // 8 MB (B, C, L)
    _Float16* qTb    = (_Float16*)(ws + 10485760);   // 8 MB (BH, L, CH)
    _Float16* kh     = (_Float16*)(ws + 18874368);   // 8 MB (BH, CH, L)
    _Float16* vTb    = (_Float16*)(ws + 27262976);   // 8 MB (BH, L, CH)
    _Float16* ah     = (_Float16*)(ws + 35651584);   // 8 MB (B, C, L)

    cvt_weights_kernel<<<(3 * CC * CC + 255) / 256, 256, 0, stream>>>(
        qkv_w, proj_w, qkvwh, projwh);

    groupnorm_kernel<<<BB * NGROUP, 256, 0, stream>>>(x, gn_w, gn_b, hn);

    qkv_gemm_kernel<<<dim3(64, 12, BB), 256, 0, stream>>>(
        qkvwh, hn, qkv_b, qTb, kh, vTb);

    attn_kernel<<<dim3(LL / 16, BB * NH), 32,
                  16 * LL * sizeof(float) + 16 * LL * sizeof(_Float16), stream>>>(
        qTb, kh, vTb, ah);

    proj_gemm_kernel<<<dim3(64, 4, BB), 256, 0, stream>>>(
        projwh, ah, proj_b, x, out);
}
